// MemTransformerLM_89919435309663
// MI455X (gfx1250) — matrix operand order repaired and checked by reading, not compiled, not run on hardware
//
// Transformer-XL (MemTransformerLM) forward on gfx1250 (MI455X).
// All GEMMs via v_wmma_f32_16x16x32_bf16 (wave32 WMMA); fp32 operands packed
// to bf16 in LDS with b128 staging. ~300 GFLOP total; weights (~200MB)
// largely resident in the 192MB L2 -> WMMA-throughput bound.
// Workspace requirement: ~220 MB (carved linearly from d_ws).

#include <hip/hip_runtime.h>
#include <hip/hip_bf16.h>
#include <math.h>

typedef __attribute__((ext_vector_type(16))) __bf16 bf16x16;
typedef __attribute__((ext_vector_type(2)))  __bf16 bf16x2;
typedef __attribute__((ext_vector_type(8)))  float   f32x8;

#define VOCAB 50257
#define LAYERS 6
#define HEADS 8
#define DHEAD 64
#define DMODEL 512
#define DINNER 2048
#define QL 512
#define ML 512
#define KL 1024
#define BS 4

// Pack two f32 into packed bf16 (RNE). Let ISel pick v_cvt_pk_bf16_f32 if the
// target has it; otherwise LLVM expands to the standard RNE bit sequence.
__device__ __forceinline__ unsigned pkbf(float a, float b) {
  bf16x2 p;
  p[0] = (__bf16)a;
  p[1] = (__bf16)b;
  return __builtin_bit_cast(unsigned, p);
}

// ---------------------------------------------------------------------------
// Generic batched GEMM: C[z][m][n] = sum_k A[z][m][k] * B[z][n][k]  (+bias,relu)
// Per-batch bases: base = (z/div)*so + (z%div)*si for A, B, C.
// btrans=1 reads B element (n,k) at base + k*ldb + n (attn @ V only).
// REQUIREMENTS (all call sites satisfy): K % 32 == 0, M % 128 == 0,
// 16B-aligned row starts (lda/ldb mult of 4, bases mult of 4).
// Block tile 128x64, K-step 32, 4 waves; each wave 64x32 = 4x2 WMMA tiles.
// ---------------------------------------------------------------------------
#define BM 128
#define BN 64
#define BK 32
#define LDSU 20          // uints per LDS row: BK/2 + 4 pad (80B, 16B aligned)
#define LDSH 40          // bf16 per LDS row

__global__ __launch_bounds__(128)
void gemm_bf16_wmma(const float* __restrict__ A, const float* __restrict__ B,
                    float* __restrict__ C,
                    int M, int N, int K, int lda, int ldb, int ldc,
                    int adiv, int aso, int asi,
                    int bdiv, int bso, int bsi,
                    int cdiv, int cso, int csi,
                    const float* __restrict__ bias, int relu, int btrans) {
  __shared__ unsigned AsU[BM * LDSU];   // 10240 B
  __shared__ unsigned BsU[BN * LDSU];   //  5120 B
  const __bf16* As = (const __bf16*)AsU;
  const __bf16* Bs = (const __bf16*)BsU;

  const int tid  = threadIdx.x;
  const int lane = tid & 31;
  const int wave = tid >> 5;
  const int wm   = (wave >> 1) * 64;     // wave M offset in tile
  const int wn   = (wave & 1) * 32;      // wave N offset in tile

  const int z = blockIdx.z;
  const long long abase = (long long)(z / adiv) * aso + (long long)(z % adiv) * asi;
  const long long bbase = (long long)(z / bdiv) * bso + (long long)(z % bdiv) * bsi;
  const long long cbase = (long long)(z / cdiv) * cso + (long long)(z % cdiv) * csi;

  const int m0 = blockIdx.y * BM;
  const int n0 = blockIdx.x * BN;

  f32x8 acc[4][2];
#pragma unroll
  for (int a = 0; a < 4; ++a)
#pragma unroll
    for (int b = 0; b < 2; ++b)
#pragma unroll
      for (int j = 0; j < 8; ++j) acc[a][b][j] = 0.f;

  const int brow = tid >> 1;           // 0..63 : B tile row this thread stages
  const int bcol = (tid & 1) * 16;     // 0/16  : 16 contiguous K elements

  const float* arow = A + abase + (long long)(m0 + tid) * lda;  // M%BM==0: valid

  for (int k0 = 0; k0 < K; k0 += BK) {
    // ---- stage A tile: 128 rows x first 16 of 32 K, one row per thread (no guard:
    //      every call site has M % 128 == 0) ----
    {
      const float4* ap = (const float4*)(arow + k0);
      const float4 v0 = ap[0], v1 = ap[1], v2 = ap[2], v3 = ap[3];
      uint4 u0, u1;
      u0.x = pkbf(v0.x, v0.y); u0.y = pkbf(v0.z, v0.w);
      u0.z = pkbf(v1.x, v1.y); u0.w = pkbf(v1.z, v1.w);
      u1.x = pkbf(v2.x, v2.y); u1.y = pkbf(v2.z, v2.w);
      u1.z = pkbf(v3.x, v3.y); u1.w = pkbf(v3.z, v3.w);
      unsigned* dst = AsU + tid * LDSU;
      ((uint4*)dst)[0] = u0;
      ((uint4*)dst)[1] = u1;
    }
    // ---- stage B tile: 64 rows, half a row per thread (only its first 8 K unless btrans) ----
    {
      unsigned* dst = BsU + brow * LDSU + (bcol >> 1);
      const int gn = n0 + brow;
      if (!btrans) {
        if (gn < N) {
          const float4* bp =
              (const float4*)(B + bbase + (long long)gn * ldb + k0 + bcol);
          const float4 v0 = bp[0], v1 = bp[1];
          uint4 u;
          u.x = pkbf(v0.x, v0.y); u.y = pkbf(v0.z, v0.w);
          u.z = pkbf(v1.x, v1.y); u.w = pkbf(v1.z, v1.w);
          ((uint4*)dst)[0] = u;
        } else {
          uint4 zz; zz.x = zz.y = zz.z = zz.w = 0u;
          ((uint4*)dst)[0] = zz;
        }
      } else {
        // strided (transposed) access: B(n,k) at bbase + k*ldb + n
        float v[16];
#pragma unroll
        for (int j = 0; j < 16; ++j)
          v[j] = (gn < N)
                     ? B[bbase + (long long)(k0 + bcol + j) * ldb + gn]
                     : 0.f;
        uint4 u;
        u.x = pkbf(v[0], v[1]);   u.y = pkbf(v[2], v[3]);
        u.z = pkbf(v[4], v[5]);   u.w = pkbf(v[6], v[7]);
        uint4 u2;
        u2.x = pkbf(v[8], v[9]);  u2.y = pkbf(v[10], v[11]);
        u2.z = pkbf(v[12], v[13]); u2.w = pkbf(v[14], v[15]);
        ((uint2*)dst)[0] = make_uint2(u.x, u.y);
        ((uint2*)dst)[1] = make_uint2(u.z, u.w);
        ((uint2*)dst)[2] = make_uint2(u2.x, u2.y);
        ((uint2*)dst)[3] = make_uint2(u2.z, u2.w);
      }
    }
    __syncthreads();

    // ---- fragments (ISA 7.12.2 layouts) + WMMA ----
    const int am   = lane & 15;
    const int aoff = (lane >> 4) << 3;   // A: upper half-wave holds K+8
    const int boff = (lane >> 4) << 3;   // B: upper half-wave holds K+8

    bf16x16 afrag[4], bfrag[2];
#pragma unroll
    for (int t = 0; t < 4; ++t) {
      const __bf16* ar = As + (wm + t * 16 + am) * LDSH;
#pragma unroll
      for (int i = 0; i < 16; ++i)
        afrag[t][i] = ar[(i < 8 ? i : i + 8) + aoff];
    }
#pragma unroll
    for (int t = 0; t < 2; ++t) {
      const __bf16* br = Bs + (wn + t * 16 + am) * LDSH;
#pragma unroll
      for (int i = 0; i < 16; ++i) bfrag[t][i] = br[(i < 8 ? i : i + 8) + boff];
    }
#pragma unroll
    for (int mi = 0; mi < 4; ++mi)
#pragma unroll
      for (int ni = 0; ni < 2; ++ni)
        acc[mi][ni] = __builtin_amdgcn_wmma_f32_16x16x32_bf16(
            false, afrag[mi], false, bfrag[ni], (short)0, acc[mi][ni],
            false, false);
    __syncthreads();
  }

  // ---- epilogue: C layout VGPR j -> M = j + 8*(lane>=16), N = lane&15 ----
  const int crow = (lane >> 4) << 3;
  const int ccol = lane & 15;
#pragma unroll
  for (int mi = 0; mi < 4; ++mi)
#pragma unroll
    for (int ni = 0; ni < 2; ++ni) {
      const int n = n0 + wn + ni * 16 + ccol;
      if (n < N) {
        const float bv = bias ? bias[n] : 0.f;
#pragma unroll
        for (int j = 0; j < 8; ++j) {
          const int m = m0 + wm + mi * 16 + crow + j;   // M%128==0: in range
          float v = acc[mi][ni][j] + bv;
          if (relu) v = v > 0.f ? v : 0.f;
          C[cbase + (long long)m * ldc + n] = v;
        }
      }
    }
}

// ---------------------------------------------------------------------------
// Elementwise / reduction helper kernels
// ---------------------------------------------------------------------------
__global__ void embed_kernel(const int* __restrict__ inp,
                             const float* __restrict__ embW,
                             float* __restrict__ h) {
  const int row = blockIdx.x;              // i*BS + b
  const int tok = inp[row];
  const float s = 22.62741699796952f;      // sqrt(512)
  for (int d = threadIdx.x; d < DMODEL; d += blockDim.x)
    h[(long long)row * DMODEL + d] = embW[(long long)tok * DMODEL + d] * s;
}

__global__ void posemb_kernel(float* __restrict__ r) {
  const int p = blockIdx.x;                // 0..KL-1
  const float pos = (float)(KL - 1 - p);
  for (int d = threadIdx.x; d < DMODEL; d += blockDim.x) {
    const int i = (d < 256) ? d : d - 256;
    const float invf = powf(10000.f, -(float)i / 256.f);
    const float a = pos * invf;
    r[p * DMODEL + d] = (d < 256) ? sinf(a) : cosf(a);
  }
}

__global__ void concat_kernel(const float* __restrict__ memsl,
                              const float* __restrict__ h,
                              float* __restrict__ cat) {
  const long long idx = (long long)blockIdx.x * blockDim.x + threadIdx.x;
  const long long memN = (long long)ML * BS * DMODEL;
  if (idx < memN) cat[idx] = memsl[idx];
  else if (idx < 2 * memN) cat[idx] = h[idx - memN];
}

__global__ void split_heads_kernel(const float* __restrict__ heads,
                                   const float* __restrict__ rwb,
                                   const float* __restrict__ rrb,
                                   float* __restrict__ kbuf,
                                   float* __restrict__ vbuf,
                                   float* __restrict__ qw,
                                   float* __restrict__ qr) {
  const long long idx = (long long)blockIdx.x * blockDim.x + threadIdx.x;
  if (idx >= (long long)KL * BS * DMODEL) return;
  const int c = (int)(idx & (DMODEL - 1));
  const long long row = idx >> 9;
  const float* hp = heads + row * (3 * DMODEL);
  kbuf[idx] = hp[DMODEL + c];
  vbuf[idx] = hp[2 * DMODEL + c];
  if (row >= (long long)ML * BS) {
    const long long q = idx - (long long)ML * BS * DMODEL;
    const float qv = hp[c];
    qw[q] = qv + rwb[c];
    qr[q] = qv + rrb[c];
  }
}

// rel-shift(BD) + AC, *scale, causal mask, row softmax. attn may alias AC.
__global__ __launch_bounds__(256)
void attn_softmax_kernel(const float* __restrict__ AC,
                         const float* __restrict__ BD,
                         float* __restrict__ attn) {
  const int i = blockIdx.x;   // query row
  const int z = blockIdx.y;   // b*HEADS + n
  const int tid = threadIdx.x;
  __shared__ float red[256];
  const float scale = 0.125f;          // 1/sqrt(DHEAD)
  const long long rbase = ((long long)z * QL + i) * KL;

  float sv[4];
#pragma unroll
  for (int u = 0; u < 4; ++u) {
    const int j = tid + u * 256;
    const float ac = AC[rbase + j];
    const long long t = (long long)QL + (long long)i * KL + j;
    const int i0 = (int)(t / (KL + 1));
    const int m = (int)(t - (long long)i0 * (KL + 1));
    const float bd = (m == 0) ? 0.f
                              : BD[((long long)z * QL + i0) * KL + (m - 1)];
    float s = (ac + bd) * scale;
    if (j > i + ML) s = -1e30f;
    sv[u] = s;
  }
  float mx = fmaxf(fmaxf(sv[0], sv[1]), fmaxf(sv[2], sv[3]));
  red[tid] = mx; __syncthreads();
  for (int o = 128; o > 0; o >>= 1) {
    if (tid < o) red[tid] = fmaxf(red[tid], red[tid + o]);
    __syncthreads();
  }
  mx = red[0]; __syncthreads();
  float sum = 0.f;
#pragma unroll
  for (int u = 0; u < 4; ++u) { sv[u] = __expf(sv[u] - mx); sum += sv[u]; }
  red[tid] = sum; __syncthreads();
  for (int o = 128; o > 0; o >>= 1) {
    if (tid < o) red[tid] += red[tid + o];
    __syncthreads();
  }
  const float inv = 1.f / red[0];
#pragma unroll
  for (int u = 0; u < 4; ++u) attn[rbase + tid + u * 256] = sv[u] * inv;
}

// h = LayerNorm(h + res) * g + b    (in place, one block per row, D=512)
__global__ __launch_bounds__(256)
void add_ln_kernel(float* __restrict__ h, const float* __restrict__ res,
                   const float* __restrict__ g, const float* __restrict__ b) {
  const int row = blockIdx.x;
  const int tid = threadIdx.x;
  __shared__ float red[256];
  const long long base = (long long)row * DMODEL;
  const float x0 = h[base + tid] + res[base + tid];
  const float x1 = h[base + tid + 256] + res[base + tid + 256];
  red[tid] = x0 + x1; __syncthreads();
  for (int o = 128; o > 0; o >>= 1) {
    if (tid < o) red[tid] += red[tid + o];
    __syncthreads();
  }
  const float mu = red[0] / 512.f; __syncthreads();
  const float d0 = x0 - mu, d1 = x1 - mu;
  red[tid] = d0 * d0 + d1 * d1; __syncthreads();
  for (int o = 128; o > 0; o >>= 1) {
    if (tid < o) red[tid] += red[tid + o];
    __syncthreads();
  }
  const float inv = rsqrtf(red[0] / 512.f + 1e-5f);
  h[base + tid]       = d0 * inv * g[tid] + b[tid];
  h[base + tid + 256] = d1 * inv * g[tid + 256] + b[tid + 256];
}

__global__ void stats_init_kernel(float* m, float* s, float* tl) {
  const int i = blockIdx.x * blockDim.x + threadIdx.x;
  if (i < QL * BS) { m[i] = -3.4e38f; s[i] = 0.f; tl[i] = 0.f; }
}

// Online log-sum-exp merge over a vocab chunk; records target logit.
__global__ __launch_bounds__(256)
void logit_chunk_kernel(const float* __restrict__ chunk, int cn, int c0,
                        const int* __restrict__ target,
                        float* __restrict__ sm, float* __restrict__ ss,
                        float* __restrict__ tl) {
  const int row = blockIdx.x;
  const int tid = threadIdx.x;
  __shared__ float red[256];
  const float* p = chunk + (long long)row * cn;
  float mx = -3.4e38f;
  for (int j = tid; j < cn; j += 256) mx = fmaxf(mx, p[j]);
  red[tid] = mx; __syncthreads();
  for (int o = 128; o > 0; o >>= 1) {
    if (tid < o) red[tid] = fmaxf(red[tid], red[tid + o]);
    __syncthreads();
  }
  mx = red[0]; __syncthreads();
  float sum = 0.f;
  for (int j = tid; j < cn; j += 256) sum += __expf(p[j] - mx);
  red[tid] = sum; __syncthreads();
  for (int o = 128; o > 0; o >>= 1) {
    if (tid < o) red[tid] += red[tid + o];
    __syncthreads();
  }
  if (tid == 0) {
    const float om = sm[row], os = ss[row];
    const float nm = fmaxf(om, mx);
    ss[row] = os * __expf(om - nm) + red[0] * __expf(mx - nm);
    sm[row] = nm;
    const int t = target[row];
    if (t >= c0 && t < c0 + cn) tl[row] = p[t - c0];
  }
}

__global__ void nll_kernel(const float* sm, const float* ss, const float* tl,
                           float* out) {
  const int i = blockIdx.x * blockDim.x + threadIdx.x;
  if (i < QL * BS) out[i] = sm[i] + __logf(ss[i]) - tl[i];
}

// ---------------------------------------------------------------------------
static void launch_gemm(hipStream_t st, const float* A, const float* B, float* C,
                        int M, int N, int K, int lda, int ldb, int ldc, int batch,
                        int adiv, int aso, int asi,
                        int bdiv, int bso, int bsi,
                        int cdiv, int cso, int csi,
                        const float* bias, int relu, int btrans) {
  dim3 grid((N + BN - 1) / BN, (M + BM - 1) / BM, batch);
  gemm_bf16_wmma<<<grid, 128, 0, st>>>(A, B, C, M, N, K, lda, ldb, ldc,
                                       adiv, aso, asi, bdiv, bso, bsi,
                                       cdiv, cso, csi, bias, relu, btrans);
}

extern "C" void kernel_launch(void* const* d_in, const int* in_sizes, int n_in,
                              void* d_out, int out_size, void* d_ws, size_t ws_size,
                              hipStream_t stream) {
  const int*   inp    = (const int*)d_in[0];
  const int*   target = (const int*)d_in[1];
  const float* mems   = (const float*)d_in[2];
  const float* embW   = (const float*)d_in[3];
  const float* outW   = (const float*)d_in[4];
  const float* outb   = (const float*)d_in[5];
  const float* rwb    = (const float*)d_in[6];
  const float* rrb    = (const float*)d_in[7];
  const float* qkvW   = (const float*)d_in[8];
  const float* rW     = (const float*)d_in[9];
  const float* oW     = (const float*)d_in[10];
  const float* ln1g   = (const float*)d_in[11];
  const float* ln1b   = (const float*)d_in[12];
  const float* ffW1   = (const float*)d_in[13];
  const float* ffb1   = (const float*)d_in[14];
  const float* ffW2   = (const float*)d_in[15];
  const float* ffb2   = (const float*)d_in[16];
  const float* ln2g   = (const float*)d_in[17];
  const float* ln2b   = (const float*)d_in[18];
  float* out = (float*)d_out;

  char* p = (char*)d_ws;
  auto carve = [&](size_t nf) { float* r = (float*)p; p += nf * sizeof(float); return r; };
  float* h     = carve((size_t)2048 * 512);
  float* cat   = carve((size_t)4096 * 512);
  float* heads = carve((size_t)4096 * 1536);
  float* kbuf  = carve((size_t)4096 * 512);
  float* vbuf  = carve((size_t)4096 * 512);
  float* qw    = carve((size_t)2048 * 512);
  float* qr    = carve((size_t)2048 * 512);
  float* rbuf  = carve((size_t)1024 * 512);
  float* rk    = carve((size_t)1024 * 512);
  float* AC    = carve((size_t)32 * 512 * 1024);
  float* BD    = carve((size_t)32 * 512 * 1024);
  float* vec   = carve((size_t)2048 * 512);
  float* core1 = carve((size_t)2048 * 2048);
  float* proj  = carve((size_t)2048 * 512);
  float* sm    = carve(2048);
  float* ss    = carve(2048);
  float* tl    = carve(2048);
  float* chunkbuf = BD;   // reuse BD region for logits chunks (after attention)

  embed_kernel<<<2048, 256, 0, stream>>>(inp, embW, h);
  posemb_kernel<<<1024, 256, 0, stream>>>(rbuf);

  for (int l = 0; l < LAYERS; ++l) {
    concat_kernel<<<8192, 256, 0, stream>>>(mems + (size_t)l * ML * BS * DMODEL, h, cat);

    // heads = cat @ qkv_W[l]^T   [4096 x 1536]
    launch_gemm(stream, cat, qkvW + (size_t)l * 1536 * 512, heads,
                4096, 1536, 512, 512, 512, 1536, 1,
                1, 0, 0, 1, 0, 0, 1, 0, 0, nullptr, 0, 0);
    split_heads_kernel<<<8192, 256, 0, stream>>>(heads, rwb, rrb, kbuf, vbuf, qw, qr);

    // rk = r @ r_W[l]^T   [1024 x 512]
    launch_gemm(stream, rbuf, rW + (size_t)l * 512 * 512, rk,
                1024, 512, 512, 512, 512, 512, 1,
                1, 0, 0, 1, 0, 0, 1, 0, 0, nullptr, 0, 0);

    // AC[z=(b,n)]: [512 x 1024] = (q+rwb) @ k^T, per-batch bases = z*64
    launch_gemm(stream, qw, kbuf, AC, 512, 1024, 64, 2048, 2048, 1024, 32,
                8, 512, 64, 32, 0, 64, 32, 0, 512 * 1024, nullptr, 0, 0);
    // BD[z]: (q+rrb) @ rk^T; rk base depends only on head: (z%8)*64
    launch_gemm(stream, qr, rk, BD, 512, 1024, 64, 2048, 512, 1024, 32,
                8, 512, 64, 8, 0, 64, 32, 0, 512 * 1024, nullptr, 0, 0);

    attn_softmax_kernel<<<dim3(QL, 32), 256, 0, stream>>>(AC, BD, AC);

    // vec[z]: [512 x 64] = attn @ v   (B transposed access)
    launch_gemm(stream, AC, vbuf, vec, 512, 64, 1024, 1024, 2048, 2048, 32,
                32, 0, 512 * 1024, 32, 0, 64, 8, 512, 64, nullptr, 0, 1);

    // proj = vec @ o_W[l]^T   [2048 x 512]
    launch_gemm(stream, vec, oW + (size_t)l * 512 * 512, proj,
                2048, 512, 512, 512, 512, 512, 1,
                1, 0, 0, 1, 0, 0, 1, 0, 0, nullptr, 0, 0);
    add_ln_kernel<<<2048, 256, 0, stream>>>(h, proj, ln1g + l * 512, ln1b + l * 512);

    // FFN
    launch_gemm(stream, h, ffW1 + (size_t)l * 2048 * 512, core1,
                2048, 2048, 512, 512, 512, 2048, 1,
                1, 0, 0, 1, 0, 0, 1, 0, 0, ffb1 + l * 2048, 1, 0);
    launch_gemm(stream, core1, ffW2 + (size_t)l * 512 * 2048, proj,
                2048, 512, 2048, 2048, 2048, 512, 1,
                1, 0, 0, 1, 0, 0, 1, 0, 0, ffb2 + l * 512, 0, 0);
    add_ln_kernel<<<2048, 256, 0, stream>>>(h, proj, ln2g + l * 512, ln2b + l * 512);
  }

  // Chunked vocab log-softmax NLL (online LSE merge; avoids 412MB logits).
  stats_init_kernel<<<8, 256, 0, stream>>>(sm, ss, tl);
  for (int c0 = 0; c0 < VOCAB; c0 += 4096) {
    const int cn = (VOCAB - c0 < 4096) ? (VOCAB - c0) : 4096;
    launch_gemm(stream, h, outW + (size_t)c0 * 512, chunkbuf,
                2048, cn, 512, 512, 512, cn, 1,
                1, 0, 0, 1, 0, 0, 1, 0, 0, outb + c0, 0, 0);
    logit_chunk_kernel<<<2048, 256, 0, stream>>>(chunkbuf, cn, c0, target, sm, ss, tl);
  }
  nll_kernel<<<8, 256, 0, stream>>>(sm, ss, tl, out);
}
